// SelfAttentionConv2d_26963804684835
// MI455X (gfx1250) — matrix operand order repaired and checked by reading, not compiled, not run on hardware
//
#include <hip/hip_runtime.h>
#include <stdint.h>

typedef __attribute__((ext_vector_type(16))) _Float16 v16h;
typedef __attribute__((ext_vector_type(8)))  _Float16 v8h;
typedef __attribute__((ext_vector_type(8)))  float    v8f;

constexpr int KSZ  = 7;
constexpr int PADW = 3;
constexpr int CIN  = 128;
constexpr int OCH  = 128;
constexpr int HIMG = 64;
constexpr int WIMG = 64;
constexpr int PH   = 70;
constexpr int PW   = 70;
constexpr int PROW = 80;           // padded row stride: 32B alignment + slack
constexpr int PPOS = PH * PROW;    // 5600 position slots per batch grid
constexpr int NCTR = HIMG * WIMG;  // 4096 centers
constexpr int NBAT = 4;

// WMMA f16 A-operand k map used in this file (contiguous k order, wave32)
__device__ __forceinline__ int kOfA(int t, int lane) {
  // 16x32 f16 A: K = t + 16*(lane/16), M = lane%16
  return t + ((lane >> 4) << 4);
}
__device__ __forceinline__ v8f zero8() {
  v8f z;
  for (int i = 0; i < 8; ++i) z[i] = 0.f;
  return z;
}
__device__ __forceinline__ v8h zero8h() {
  v8h z;
  for (int i = 0; i < 8; ++i) z[i] = (_Float16)0.f;
  return z;
}

// ---------------------------------------------------------------------------
// Kernel 0: convert wq/wk/wv once into f16 WMMA A-fragment layout:
//   wfrag[((mat*8 + mtile)*4 + kk)*512 + lane*16 + t]
// so the projection kernel's weight path is pure aligned v16h loads.
// ---------------------------------------------------------------------------
__global__ __launch_bounds__(256)
void wfrag_kernel(const float* __restrict__ wq,
                  const float* __restrict__ wk,
                  const float* __restrict__ wv,
                  _Float16* __restrict__ wfrag) {
  const int bx    = blockIdx.x;      // 0..23 = mat*8 + mtile
  const int mat   = bx >> 3;
  const int mtile = bx & 7;
  const float* w  = (mat == 0) ? wq : (mat == 1) ? wk : wv;
  const int tid   = threadIdx.x;
  for (int e = 0; e < 8; ++e) {
    int idx = tid + e * 256;         // 0..2047 = kk*512 + lane*16 + t
    int kk  = idx >> 9;
    int l2  = (idx >> 4) & 31;
    int t   = idx & 15;
    int row = mtile * 16 + (l2 & 15);
    int c   = kk * 32 + kOfA(t, l2);
    wfrag[(size_t)bx * 2048 + idx] = (_Float16)w[row * CIN + c];
  }
}

// ---------------------------------------------------------------------------
// Kernel 1: q/k/v projections, one padded row (y) per block. No div/mod.
//   qhT[b][ctr][c]   f16 centers, channel-contiguous
//   khT[b][pos80][c] f16 channel-contiguous
//   vh [b][c][pos80] f16 column-contiguous
// 8 waves: wave -> 16 output channels. Prebuilt weight fragments loaded once
// per block (12 x v16h per lane) and reused across the row's 5 x-tiles.
// ---------------------------------------------------------------------------
__global__ __launch_bounds__(256)
void qkv_proj_kernel(const float* __restrict__ xin,
                     const _Float16* __restrict__ wfrag,
                     _Float16* __restrict__ qhT,
                     _Float16* __restrict__ khT,
                     _Float16* __restrict__ vh) {
  __shared__ _Float16 ldsRow[PROW * 128];  // [xx][c]

  const int tid   = threadIdx.x;
  const int lane  = tid & 31;
  const int wave  = tid >> 5;
  const int s     = lane >> 4;
  const int nIdx  = lane & 15;
  const int y     = blockIdx.x;   // padded row 0..69
  const int b     = blockIdx.y;
  const int yy    = y - PADW;
  const int mBase = wave * 16;

  if (yy < 0 || yy >= HIMG) {
    // Zero-input row: projections are exactly zero. Store zeros, done.
    v8h z = zero8h();
    for (int e = 0; e < 5; ++e) {
      int off = (tid + e * 256) * 8;   // 10240 halves of the khT row
      *(v8h*)&khT[((size_t)b * PPOS + y * PROW) * 128 + off] = z;
    }
    {
      int c    = tid >> 1;
      int part = tid & 1;
      for (int e = 0; e < 5; ++e)
        *(v8h*)&vh[(size_t)(b * OCH + c) * PPOS + y * PROW + part * 40 + e * 8] = z;
    }
    return;
  }

  // Stage row: zero border columns (0..2, 67..79)
  for (int e = 0; e < 8; ++e) {
    int idx    = tid + e * 256;          // 0..2047 = colIdx*128 + c
    int colIdx = idx >> 7;               // 0..15
    int c      = idx & 127;
    int xx     = (colIdx < 3) ? colIdx : (64 + colIdx);
    ldsRow[xx * 128 + c] = (_Float16)0.f;
  }
  // Copy interior columns 3..66 with coalesced float4 reads
  for (int e = 0; e < 8; ++e) {
    int idx = tid + e * 256;             // 0..2047
    int c   = idx >> 4;                  // 0..127
    int j4  = (idx & 15) * 4;            // 0..60
    const float4 f =
        *(const float4*)&xin[((size_t)(b * CIN + c) * HIMG + yy) * WIMG + j4];
    ldsRow[(j4 + 3) * 128 + c] = (_Float16)f.x;
    ldsRow[(j4 + 4) * 128 + c] = (_Float16)f.y;
    ldsRow[(j4 + 5) * 128 + c] = (_Float16)f.z;
    ldsRow[(j4 + 6) * 128 + c] = (_Float16)f.w;
  }
  __syncthreads();

  // Prebuilt weight fragments: 12 aligned 32B loads, no conversion VALU
  v16h wfq[4], wfk[4], wfv[4];
  #pragma unroll
  for (int kk = 0; kk < 4; ++kk) {
    wfq[kk] = *(const v16h*)&wfrag[(size_t)(((0 * 8 + wave) * 4 + kk) * 512) + lane * 16];
    wfk[kk] = *(const v16h*)&wfrag[(size_t)(((1 * 8 + wave) * 4 + kk) * 512) + lane * 16];
    wfv[kk] = *(const v16h*)&wfrag[(size_t)(((2 * 8 + wave) * 4 + kk) * 512) + lane * 16];
  }

  const bool ctrRow = (y >= PADW && y < PH - PADW);
  for (int nt = 0; nt < 5; ++nt) {
    v8f aq = zero8(), ak = zero8(), av = zero8();
    const int xx = nt * 16 + nIdx;       // B lane column (<=79, LDS zero-padded)
    #pragma unroll
    for (int kk = 0; kk < 4; ++kk) {
      v16h bf = *(const v16h*)&ldsRow[xx * 128 + kk * 32 + 16 * s];
      aq = __builtin_amdgcn_wmma_f32_16x16x32_f16(false, wfq[kk], false, bf, (short)0, aq, false, false);
      ak = __builtin_amdgcn_wmma_f32_16x16x32_f16(false, wfk[kk], false, bf, (short)0, ak, false, false);
      av = __builtin_amdgcn_wmma_f32_16x16x32_f16(false, wfv[kk], false, bf, (short)0, av, false, false);
    }
    if (xx < PW) {
      int pos80  = y * PROW + xx;
      int ocBase = mBase + 8 * s;        // D: M = r + 8*s -> 8 consecutive ch
      v8h pq, pk, pv;
      #pragma unroll
      for (int r = 0; r < 8; ++r) {
        pq[r] = (_Float16)aq[r];
        pk[r] = (_Float16)ak[r];
        pv[r] = (_Float16)av[r];
      }
      *(v8h*)&khT[((size_t)b * PPOS + pos80) * 128 + ocBase] = pk;
      #pragma unroll
      for (int r = 0; r < 8; ++r)
        vh[(size_t)(b * OCH + ocBase + r) * PPOS + pos80] = pv[r];
      if (ctrRow && xx >= PADW && xx < PW - PADW)
        *(v8h*)&qhT[((size_t)b * NCTR + (y - PADW) * WIMG + (xx - PADW)) * 128 + ocBase] = pq;
    }
  }
}

// ---------------------------------------------------------------------------
// Kernel 2: fused local attention per (batch, row h, 16-center tile).
// ---------------------------------------------------------------------------
__global__ __launch_bounds__(256)
void attn_kernel(const _Float16* __restrict__ qhT,
                 const _Float16* __restrict__ khT,
                 const _Float16* __restrict__ vh,
                 const float* __restrict__ relx,
                 const float* __restrict__ rely,
                 float* __restrict__ out) {
  __shared__ _Float16 ldsQ[16 * 128];     // [m][c]
  __shared__ float    ldsLog[16 * 52];    // logits [m][ki*7+kj]
  __shared__ float    ldsAttn[16 * 52];   // softmax result
  __shared__ float    ldsBX[16 * 8];      // bias_x[m][kj]
  __shared__ float    ldsBY[16 * 8];      // bias_y[m][ki]
  __shared__ _Float16 ldsAb[7 * 32 * 16]; // banded attn fragments [ki][lane][t]

  const int tid   = threadIdx.x;
  const int lane  = tid & 31;
  const int wave  = tid >> 5;
  const int s     = lane >> 4;
  const int n     = lane & 15;
  const int wBase = blockIdx.x * 16;
  const int h     = blockIdx.y;
  const int b     = blockIdx.z;

  // Early prefetch of the k-row / v-row this block is about to consume;
  // overlaps the q staging and barrier below (global_prefetch_b8).
  if (wave < 7)
    __builtin_prefetch(&khT[((size_t)b * PPOS + (h + wave) * PROW + wBase + n) * 128], 0, 1);
  __builtin_prefetch(&vh[(size_t)(b * OCH + wave * 16 + n) * PPOS + h * PROW + wBase], 0, 1);

  // Phase 0: stage q tile 16 centers x 128 channels (v8h copies)
  {
    int m  = tid >> 4;
    int c0 = (tid & 15) * 8;
    *(v8h*)&ldsQ[m * 128 + c0] =
        *(const v8h*)&qhT[((size_t)b * NCTR + h * WIMG + wBase + m) * 128 + c0];
  }
  __syncthreads();

  // Phase 1 (wave-uniform branch: EXEC all-ones inside for WMMA)
  if (wave < 7) {
    const int ki  = wave;
    const int row = h + ki;
    v8f s0 = zero8(), s1 = zero8();
    for (int kk = 0; kk < 4; ++kk) {
      v8h lo = *(const v8h*)&ldsQ[n * 128 + kk * 32 + 16 * s];
      v8h hi = *(const v8h*)&ldsQ[n * 128 + kk * 32 + 16 * s + 8];
      v16h af = __builtin_shufflevector(lo, hi, 0, 1, 2, 3, 4, 5, 6, 7,
                                        8, 9, 10, 11, 12, 13, 14, 15);
      // cols >= 70 hit in-bounds slack; those S columns are never extracted.
      size_t base = ((size_t)b * PPOS + row * PROW + wBase + n) * 128 + kk * 32 + 16 * s;
      v16h b0 = *(const v16h*)&khT[base];
      v16h b1 = *(const v16h*)&khT[base + (size_t)16 * 128];
      s0 = __builtin_amdgcn_wmma_f32_16x16x32_f16(false, af, false, b0, (short)0, s0, false, false);
      s1 = __builtin_amdgcn_wmma_f32_16x16x32_f16(false, af, false, b1, (short)0, s1, false, false);
    }
    // band extraction: logits[m][ki*7+kj] = S[m][m+kj]
    #pragma unroll
    for (int nt = 0; nt < 2; ++nt) {
      #pragma unroll
      for (int r = 0; r < 8; ++r) {
        int m  = r + 8 * s;
        int ng = nt * 16 + n;
        int d  = ng - m;
        if ((unsigned)d < 7u)
          ldsLog[m * 52 + ki * 7 + d] = nt ? s1[r] : s0[r];
      }
    }
  } else {
    // rel-pos biases; channel j=c%16: j<8 -> rel_x(kj) else rel_y(ki).
    // 8-channel v8h chunks alternate parity: even chunk -> rel_x, odd -> rel_y.
    if (lane < 16) {
      int   m = lane;
      float bx[7] = {0, 0, 0, 0, 0, 0, 0};
      float by[7] = {0, 0, 0, 0, 0, 0, 0};
      for (int c8 = 0; c8 < 16; c8 += 2) {
        v8h qx8 = *(const v8h*)&ldsQ[m * 128 + c8 * 8];
        v8h qy8 = *(const v8h*)&ldsQ[m * 128 + c8 * 8 + 8];
        #pragma unroll
        for (int r = 0; r < 8; ++r) {
          float qx = (float)qx8[r], qy = (float)qy8[r];
          #pragma unroll
          for (int t = 0; t < 7; ++t) {
            bx[t] += qx * relx[r * KSZ + t];
            by[t] += qy * rely[r * KSZ + t];
          }
        }
      }
      #pragma unroll
      for (int t = 0; t < 7; ++t) {
        ldsBX[m * 8 + t] = bx[t];
        ldsBY[m * 8 + t] = by[t];
      }
    }
  }
  __syncthreads();

  // Phase 2: 49-way softmax per center
  if (tid < 16) {
    int   m  = tid;
    float mx = -3.0e38f;
    for (int i = 0; i < 49; ++i) {
      int   ki2 = i / 7, kj = i % 7;
      float v = ldsLog[m * 52 + i] + ldsBX[m * 8 + kj] + ldsBY[m * 8 + ki2];
      ldsAttn[m * 52 + i] = v;
      mx = fmaxf(mx, v);
    }
    float sum = 0.f;
    for (int i = 0; i < 49; ++i) {
      float e = __expf(ldsAttn[m * 52 + i] - mx);
      ldsAttn[m * 52 + i] = e;
      sum += e;
    }
    float inv = 1.f / sum;
    for (int i = 0; i < 49; ++i) ldsAttn[m * 52 + i] *= inv;
  }
  __syncthreads();

  // Phase 2.5: banded attn A fragments, shared by all 8 waves
  for (int e = 0; e < 14; ++e) {
    int idx = tid + e * 256;        // 0..3583 = ki*512 + lane*16 + t
    int ki  = idx >> 9;
    int l2  = (idx >> 4) & 31;
    int t   = idx & 15;
    int m   = l2 & 15;
    int x   = kOfA(t, l2);
    int d   = x - m;
    ldsAb[idx] = ((unsigned)d < 7u) ? (_Float16)ldsAttn[m * 52 + ki * 7 + d]
                                    : (_Float16)0.f;
  }
  __syncthreads();

  // Phase 3: out[m][c] += Aband_ki(16x32) x vrow_ki(32x16ch)
  {
    const int c0 = wave * 16;
    v8f acc = zero8();
    for (int ki = 0; ki < 7; ++ki) {
      v16h af = *(const v16h*)&ldsAb[(ki * 32 + lane) * 16];
      v16h bv = *(const v16h*)&vh[(size_t)(b * OCH + c0 + n) * PPOS +
                                  (h + ki) * PROW + wBase + 16 * s];
      acc = __builtin_amdgcn_wmma_f32_16x16x32_f16(false, af, false, bv, (short)0, acc, false, false);
    }
    #pragma unroll
    for (int r = 0; r < 8; ++r) {
      int c = c0 + n;        // D: N -> channel
      int m = r + 8 * s;     // D: M -> center
      out[((b * OCH + c) * HIMG + h) * WIMG + wBase + m] = acc[r];
    }
  }
}

extern "C" void kernel_launch(void* const* d_in, const int* in_sizes, int n_in,
                              void* d_out, int out_size, void* d_ws, size_t ws_size,
                              hipStream_t stream) {
  (void)in_sizes; (void)n_in; (void)out_size; (void)ws_size;
  const float* xin = (const float*)d_in[0];
  const float* wq  = (const float*)d_in[1];
  const float* wk  = (const float*)d_in[2];
  const float* wv  = (const float*)d_in[3];
  const float* rx  = (const float*)d_in[4];
  const float* ry  = (const float*)d_in[5];
  float* out = (float*)d_out;

  // Workspace (f16): qhT 4MB | khT 5.7MB | vh 5.7MB | wfrag 96KB
  _Float16* qhT   = (_Float16*)d_ws;
  _Float16* khT   = qhT + (size_t)NBAT * NCTR * 128;
  _Float16* vh    = khT + (size_t)NBAT * PPOS * 128;
  _Float16* wfrag = vh + (size_t)NBAT * OCH * PPOS;

  wfrag_kernel<<<dim3(24), 256, 0, stream>>>(wq, wk, wv, wfrag);
  qkv_proj_kernel<<<dim3(PH, NBAT), 256, 0, stream>>>(xin, wfrag, qhT, khT, vh);
  attn_kernel<<<dim3(WIMG / 16, HIMG, NBAT), 256, 0, stream>>>(qhT, khT, vh, rx, ry, out);
}
